// SimpleSparseAttention_78735340471008
// MI455X (gfx1250) — matrix operand order repaired and checked by reading, compiled, not run on hardware
//
#include <hip/hip_runtime.h>
#include <hip/hip_bf16.h>
#include <stdint.h>

#define DEV __device__ __forceinline__

typedef __attribute__((ext_vector_type(16))) __bf16       bh16;
typedef __attribute__((ext_vector_type(8)))  float        v8f;
typedef __attribute__((ext_vector_type(4)))  float        f32x4;
typedef __attribute__((ext_vector_type(4)))  unsigned int u32x4;

union FragU { u32x4 u[2]; bh16 v; };

constexpr int N_  = 1024;
constexpr int D_  = 1024;
constexpr int H_  = 16;
constexpr int HD_ = 64;
constexpr int C_  = 64;
constexpr int G_  = 16;

DEV float bf2f(unsigned short u) { return __uint_as_float(((unsigned)u) << 16); }
DEV unsigned short f2bf(float f) {
  unsigned u = __float_as_uint(f);
  unsigned r = u + 0x7FFFu + ((u >> 16) & 1u);
  return (unsigned short)(r >> 16);
}
// unpack a dword holding two bf16 (little-endian: lo = first element)
DEV void bf2x(unsigned u, float& lo, float& hi) {
  lo = __uint_as_float(u << 16);
  hi = __uint_as_float(u & 0xFFFF0000u);
}
DEV unsigned pack2bf(float a, float b) {
  return (unsigned)f2bf(a) | ((unsigned)f2bf(b) << 16);
}

// ---- WMMA fragment loaders (ISA 7.12.2, bf16, wave32) ----
DEV bh16 load_a_frag(const unsigned short* base, int stride, int row0, int k0) {
  int lane = threadIdx.x & 31;
  int hf = lane >> 4, r = lane & 15;
  const unsigned short* p = base + (size_t)(row0 + r) * stride + k0 + hf * 8;
  FragU f;
  f.u[0] = *(const u32x4*)(p);
  f.u[1] = *(const u32x4*)(p + 16);
  return f.v;
}
DEV bh16 load_b_frag(const unsigned short* base, int stride, int col0, int k0) {
  int lane = threadIdx.x & 31;
  int hf = lane >> 4, n = lane & 15;
  const unsigned short* p = base + (size_t)(col0 + n) * stride + k0 + hf * 8;
  FragU f;
  f.u[0] = *(const u32x4*)(p);
  f.u[1] = *(const u32x4*)(p + 16);
  return f.v;
}

// ---------------- f32 -> bf16 conversion, 8 elems/thread ----------------
__global__ void ssa_cvt_bf16x8(const float* __restrict__ in, unsigned short* __restrict__ out, int n8) {
  int i = blockIdx.x * blockDim.x + threadIdx.x;
  if (i >= n8) return;
  const f32x4* ip = (const f32x4*)(in) + i * 2;
  f32x4 a = ip[0], b = ip[1];
  u32x4 o;
  o[0] = pack2bf(a[0], a[1]);
  o[1] = pack2bf(a[2], a[3]);
  o[2] = pack2bf(b[0], b[1]);
  o[3] = pack2bf(b[2], b[3]);
  *((u32x4*)(out) + i) = o;
}

// ---------------- GEMM: Y(MxN) = A(MxK) * B(NxK)^T, one wave per 32x32 tile ----------------
template <bool BF16OUT>
__global__ __launch_bounds__(32) void ssa_gemm_wmma(const unsigned short* __restrict__ A,
                                                    const unsigned short* __restrict__ B,
                                                    void* __restrict__ Cout,
                                                    int N, int K) {
  int n0 = blockIdx.x * 32;
  int m0 = blockIdx.y * 32;
  int lane = threadIdx.x & 31;
  v8f c[2][2] = {};
  for (int kk = 0; kk < K; kk += 32) {
    if (kk + 32 < K) {
      __builtin_prefetch(A + (size_t)(m0 + lane) * K + kk + 32, 0, 0);
      __builtin_prefetch(B + (size_t)(n0 + lane) * K + kk + 32, 0, 0);
    }
    bh16 a[2], b[2];
    a[0] = load_a_frag(A, K, m0,      kk);
    a[1] = load_a_frag(A, K, m0 + 16, kk);
    b[0] = load_b_frag(B, K, n0,      kk);
    b[1] = load_b_frag(B, K, n0 + 16, kk);
#pragma unroll
    for (int i = 0; i < 2; i++)
#pragma unroll
      for (int j = 0; j < 2; j++)
        c[i][j] = __builtin_amdgcn_wmma_f32_16x16x32_bf16(false, a[i], false, b[j],
                                                          (short)0, c[i][j], false, false);
  }
  int hf = lane >> 4, n = lane & 15;
#pragma unroll
  for (int i = 0; i < 2; i++)
#pragma unroll
    for (int j = 0; j < 2; j++)
#pragma unroll
      for (int r = 0; r < 8; r++) {
        int row = m0 + i * 16 + hf * 8 + r;
        int col = n0 + j * 16 + n;
        float v = c[i][j][r];
        if (BF16OUT)
          ((unsigned short*)Cout)[(size_t)row * N + col] = f2bf(v);
        else
          ((float*)Cout)[(size_t)row * N + col] = v;
      }
}

// ---------------- gate softmax over pairs ----------------
__global__ void ssa_gate(const float* __restrict__ glog, float* __restrict__ gate) {
  int i = blockIdx.x * blockDim.x + threadIdx.x;  // over N*H
  if (i >= N_ * H_) return;
  float a = glog[i * 2], b = glog[i * 2 + 1];
  float m = fmaxf(a, b);
  float ea = __expf(a - m), eb = __expf(b - m);
  float inv = 1.0f / (ea + eb);
  gate[i * 2] = ea * inv;
  gate[i * 2 + 1] = eb * inv;
}

// ---------------- k_compress: mean over chunk, 2 dims/thread ----------------
__global__ void ssa_compress(const unsigned short* __restrict__ kb, float* __restrict__ kcomp) {
  int i = blockIdx.x * blockDim.x + threadIdx.x;  // over G * D/2
  if (i >= G_ * (D_ / 2)) return;
  int g = i / (D_ / 2);
  int d0 = (i % (D_ / 2)) * 2;
  float sl = 0.0f, sh = 0.0f;
#pragma unroll 8
  for (int c = 0; c < C_; c++) {
    unsigned u = *(const unsigned*)(kb + (size_t)(g * C_ + c) * D_ + d0);
    float lo, hi; bf2x(u, lo, hi);
    sl += lo; sh += hi;
  }
  kcomp[g * D_ + d0]     = sl * (1.0f / C_);
  kcomp[g * D_ + d0 + 1] = sh * (1.0f / C_);
}

// ---------------- masked block scores + top-2 ----------------
__global__ void ssa_topk(const unsigned short* __restrict__ qb, const float* __restrict__ kcomp,
                         int* __restrict__ topk) {
  int i = blockIdx.x * blockDim.x + threadIdx.x;  // over H*N
  if (i >= H_ * N_) return;
  int h = i >> 10, n = i & (N_ - 1);
  const float NEG = -__builtin_inff();
  // hoist q row into registers (64 f32)
  float qr[HD_];
  const u32x4* qp = (const u32x4*)(qb + (size_t)n * D_ + h * HD_);
#pragma unroll
  for (int t = 0; t < 8; t++) {
    u32x4 qv = qp[t];
#pragma unroll
    for (int e = 0; e < 4; e++) bf2x(qv[e], qr[t * 8 + e * 2], qr[t * 8 + e * 2 + 1]);
  }
  float s[G_];
#pragma unroll
  for (int g = 0; g < G_; g++) {
    if (n >= C_ * g) { s[g] = NEG; continue; }
    const f32x4* kc = (const f32x4*)(kcomp + g * D_ + h * HD_);
    float acc = 0.0f;
#pragma unroll
    for (int t = 0; t < 16; t++) {
      f32x4 kv = kc[t];
      acc += qr[t * 4] * kv[0] + qr[t * 4 + 1] * kv[1] + qr[t * 4 + 2] * kv[2] + qr[t * 4 + 3] * kv[3];
    }
    s[g] = acc;
  }
  // top-2 with earliest-index tie-break (matches jax.lax.top_k)
  int b0 = 0; float v0 = s[0];
#pragma unroll
  for (int g = 1; g < G_; g++) if (s[g] > v0) { v0 = s[g]; b0 = g; }
  int b1 = (b0 == 0) ? 1 : 0; float v1 = s[b1];
#pragma unroll
  for (int g = 0; g < G_; g++) if (g != b0 && s[g] > v1) { v1 = s[g]; b1 = g; }
  topk[i * 2] = b0;
  topk[i * 2 + 1] = b1;
}

// ---------------- inter attention: one wave per (h, n), 128 gathered keys ----------------
__global__ __launch_bounds__(128) void ssa_inter(const unsigned short* __restrict__ qb,
                                                 const unsigned short* __restrict__ kb,
                                                 const unsigned short* __restrict__ vb,
                                                 const int* __restrict__ topk,
                                                 const float* __restrict__ gate,
                                                 float* __restrict__ o) {
  __shared__ float qs[4][HD_];
  __shared__ float ps[4][128];
  int w = threadIdx.x >> 5, lane = threadIdx.x & 31;
  int n = blockIdx.x * 4 + w;
  int h = blockIdx.y;
  int i0 = topk[(h * N_ + n) * 2];
  int i1 = topk[(h * N_ + n) * 2 + 1];
  qs[w][lane]      = bf2f(qb[(size_t)n * D_ + h * HD_ + lane]);
  qs[w][lane + 32] = bf2f(qb[(size_t)n * D_ + h * HD_ + lane + 32]);
  __syncthreads();
  const float scale = 0.125f;  // hd^-0.5
  float l[4];
  float m = -__builtin_inff();
#pragma unroll
  for (int j = 0; j < 4; j++) {
    int key = lane + 32 * j;
    int blk = (key < 64) ? i0 : i1;
    int row = blk * C_ + (key & 63);
    const u32x4* kp = (const u32x4*)(kb + (size_t)row * D_ + h * HD_);
    float acc = 0.0f;
#pragma unroll
    for (int t = 0; t < 8; t++) {
      u32x4 kv = kp[t];
#pragma unroll
      for (int e = 0; e < 4; e++) {
        float lo, hi; bf2x(kv[e], lo, hi);
        acc += qs[w][t * 8 + e * 2] * lo + qs[w][t * 8 + e * 2 + 1] * hi;
      }
    }
    l[j] = acc * scale;
    m = fmaxf(m, l[j]);
  }
#pragma unroll
  for (int off = 16; off; off >>= 1) m = fmaxf(m, __shfl_xor(m, off, 32));
  float ssum = 0.0f;
#pragma unroll
  for (int j = 0; j < 4; j++) { l[j] = __expf(l[j] - m); ssum += l[j]; }
#pragma unroll
  for (int off = 16; off; off >>= 1) ssum += __shfl_xor(ssum, off, 32);
  float inv = 1.0f / ssum;
#pragma unroll
  for (int j = 0; j < 4; j++) ps[w][lane + 32 * j] = l[j] * inv;
  __syncthreads();
  float g0 = gate[(n * H_ + h) * 2];
  int d0 = lane * 2;
  float a0 = 0.0f, a1 = 0.0f;
  for (int key = 0; key < 128; key++) {
    int blk = (key < 64) ? i0 : i1;
    int row = blk * C_ + (key & 63);
    float p = ps[w][key];
    unsigned uv = *(const unsigned*)(vb + (size_t)row * D_ + h * HD_ + d0);
    float lo, hi; bf2x(uv, lo, hi);
    a0 += p * lo;
    a1 += p * hi;
  }
  o[(size_t)n * D_ + h * HD_ + d0]     = g0 * a0;
  o[(size_t)n * D_ + h * HD_ + d0 + 1] = g0 * a1;
}

// ---------------- intra attention: one block per (g, h), logits via WMMA ----------------
__global__ __launch_bounds__(128) void ssa_intra(const unsigned short* __restrict__ qb,
                                                 const unsigned short* __restrict__ kb,
                                                 const unsigned short* __restrict__ vb,
                                                 const float* __restrict__ gate,
                                                 float* __restrict__ o) {
  __shared__ float l[C_][C_ + 1];
  int g = blockIdx.x, h = blockIdx.y;
  int w = threadIdx.x >> 5, lane = threadIdx.x & 31;
  int hf = lane >> 4, n16 = lane & 15;
  const float scale = 0.125f;
  const float NEG = -__builtin_inff();

  v8f c4[4] = {};
  int qrow0 = g * C_ + w * 16;
#pragma unroll
  for (int kk = 0; kk < HD_; kk += 32) {
    bh16 a = load_a_frag(qb, D_, qrow0, h * HD_ + kk);
#pragma unroll
    for (int j = 0; j < 4; j++) {
      bh16 b = load_b_frag(kb, D_, g * C_ + j * 16, h * HD_ + kk);
      c4[j] = __builtin_amdgcn_wmma_f32_16x16x32_bf16(false, a, false, b,
                                                      (short)0, c4[j], false, false);
    }
  }
#pragma unroll
  for (int j = 0; j < 4; j++)
#pragma unroll
    for (int r = 0; r < 8; r++) {
      int row = w * 16 + hf * 8 + r;
      int col = j * 16 + n16;
      float v = c4[j][r] * scale;
      l[row][col] = (col <= row) ? v : NEG;
    }
  __syncthreads();

  int row = threadIdx.x & 63;
  int dhalf = threadIdx.x >> 6;
  float m = NEG;
  for (int j = 0; j <= row; j++) m = fmaxf(m, l[row][j]);
  float s = 0.0f;
  for (int j = 0; j <= row; j++) s += __expf(l[row][j] - m);
  float inv = 1.0f / s;
  __syncthreads();
  if (dhalf == 0)
    for (int j = 0; j <= row; j++) l[row][j] = __expf(l[row][j] - m) * inv;
  __syncthreads();

  float acc[32];
#pragma unroll
  for (int d = 0; d < 32; d++) acc[d] = 0.0f;
  for (int j = 0; j <= row; j++) {
    float p = l[row][j];
    const u32x4* vp = (const u32x4*)(vb + (size_t)(g * C_ + j) * D_ + h * HD_ + dhalf * 32);
#pragma unroll
    for (int t = 0; t < 4; t++) {
      u32x4 vv = vp[t];
#pragma unroll
      for (int e = 0; e < 4; e++) {
        float lo, hi; bf2x(vv[e], lo, hi);
        acc[t * 8 + e * 2]     += p * lo;
        acc[t * 8 + e * 2 + 1] += p * hi;
      }
    }
  }
  float g1 = gate[((g * C_ + row) * H_ + h) * 2 + 1];
  float* orow = o + (size_t)(g * C_ + row) * D_ + h * HD_ + dhalf * 32;
#pragma unroll
  for (int d = 0; d < 32; d++) orow[d] += g1 * acc[d];
}

// ---------------- launch ----------------
extern "C" void kernel_launch(void* const* d_in, const int* in_sizes, int n_in,
                              void* d_out, int out_size, void* d_ws, size_t ws_size,
                              hipStream_t stream) {
  (void)in_sizes; (void)n_in; (void)out_size; (void)ws_size;
  const float* x  = (const float*)d_in[0];
  const float* Wq = (const float*)d_in[1];
  const float* Wk = (const float*)d_in[2];
  const float* Wv = (const float*)d_in[3];
  const float* Wg = (const float*)d_in[4];
  const float* Wo = (const float*)d_in[5];
  float* out = (float*)d_out;

  char* ws = (char*)d_ws;
  const size_t MB = 1u << 20;
  unsigned short* xb   = (unsigned short*)(ws + 0 * MB);
  unsigned short* Wqb  = (unsigned short*)(ws + 2 * MB);
  unsigned short* Wkb  = (unsigned short*)(ws + 4 * MB);
  unsigned short* Wvb  = (unsigned short*)(ws + 6 * MB);
  unsigned short* Wob  = (unsigned short*)(ws + 8 * MB);
  unsigned short* Wgb  = (unsigned short*)(ws + 10 * MB);
  unsigned short* qb   = (unsigned short*)(ws + 11 * MB);
  unsigned short* kb   = (unsigned short*)(ws + 13 * MB);
  unsigned short* vb   = (unsigned short*)(ws + 15 * MB);
  float*          glog = (float*)(ws + 17 * MB);
  float*          gate = (float*)(ws + 18 * MB);
  float*          kcomp= (float*)(ws + 19 * MB);
  int*            topk = (int*)(ws + 20 * MB);
  float*          obuf = (float*)(ws + 21 * MB);
  unsigned short* ob   = (unsigned short*)(ws + 25 * MB);

  const int ND = N_ * D_;  // 1M
  // f32 -> bf16 (8 elems per thread)
  ssa_cvt_bf16x8<<<(ND / 8 + 255) / 256, 256, 0, stream>>>(x,  xb,  ND / 8);
  ssa_cvt_bf16x8<<<(ND / 8 + 255) / 256, 256, 0, stream>>>(Wq, Wqb, ND / 8);
  ssa_cvt_bf16x8<<<(ND / 8 + 255) / 256, 256, 0, stream>>>(Wk, Wkb, ND / 8);
  ssa_cvt_bf16x8<<<(ND / 8 + 255) / 256, 256, 0, stream>>>(Wv, Wvb, ND / 8);
  ssa_cvt_bf16x8<<<(ND / 8 + 255) / 256, 256, 0, stream>>>(Wo, Wob, ND / 8);
  ssa_cvt_bf16x8<<<(2 * H_ * D_ / 8 + 255) / 256, 256, 0, stream>>>(Wg, Wgb, 2 * H_ * D_ / 8);

  // projections (WMMA)
  dim3 gproj(D_ / 32, N_ / 32);
  ssa_gemm_wmma<true ><<<gproj, 32, 0, stream>>>(xb, Wqb, qb, D_, D_);
  ssa_gemm_wmma<true ><<<gproj, 32, 0, stream>>>(xb, Wkb, kb, D_, D_);
  ssa_gemm_wmma<true ><<<gproj, 32, 0, stream>>>(xb, Wvb, vb, D_, D_);
  dim3 ggate(1, N_ / 32);
  ssa_gemm_wmma<false><<<ggate, 32, 0, stream>>>(xb, Wgb, glog, 2 * H_, D_);

  ssa_gate<<<(N_ * H_ + 255) / 256, 256, 0, stream>>>(glog, gate);
  ssa_compress<<<(G_ * D_ / 2 + 255) / 256, 256, 0, stream>>>(kb, kcomp);
  ssa_topk<<<(H_ * N_ + 255) / 256, 256, 0, stream>>>(qb, kcomp, topk);

  dim3 ginter(N_ / 4, H_);
  ssa_inter<<<ginter, 128, 0, stream>>>(qb, kb, vb, topk, gate, obuf);
  dim3 gintra(G_, H_);
  ssa_intra<<<gintra, 128, 0, stream>>>(qb, kb, vb, gate, obuf);

  // blend -> bf16, output projection (WMMA)
  ssa_cvt_bf16x8<<<(ND / 8 + 255) / 256, 256, 0, stream>>>(obuf, ob, ND / 8);
  ssa_gemm_wmma<false><<<gproj, 32, 0, stream>>>(ob, Wob, out, D_, D_);
}
